// MaskedMHA_9259949490961
// MI455X (gfx1250) — matrix operand order repaired and checked by reading, not compiled, not run on hardware
//
#include <hip/hip_runtime.h>

typedef __attribute__((ext_vector_type(16))) _Float16 v16h;
typedef __attribute__((ext_vector_type(8)))  _Float16 v8h;
typedef __attribute__((ext_vector_type(2)))  _Float16 v2h;
typedef __attribute__((ext_vector_type(8)))  float    v8f;

#define B_ 4
#define H_ 8
#define L_ 2048
#define D_ 64
#define MODEL_ (H_ * D_)
#define BH_ (B_ * H_)

// ---------------------------------------------------------------------------
// Pre-pass 1: elementwise fp32 -> f16 (K, row-major [bh][L][D])
// ---------------------------------------------------------------------------
__global__ __launch_bounds__(256)
void conv_f16(const float* __restrict__ in, _Float16* __restrict__ out) {
    const size_t i = ((size_t)blockIdx.x * 256 + threadIdx.x) * 8;
    float4 f0 = *(const float4*)(in + i);
    float4 f1 = *(const float4*)(in + i + 4);
    v8h o;
    o[0] = (_Float16)f0.x; o[1] = (_Float16)f0.y;
    o[2] = (_Float16)f0.z; o[3] = (_Float16)f0.w;
    o[4] = (_Float16)f1.x; o[5] = (_Float16)f1.y;
    o[6] = (_Float16)f1.z; o[7] = (_Float16)f1.w;
    *(v8h*)(out + i) = o;
}

// key permutation inside a 64-key block (P is stored to LDS as packed pairs;
// V's key dim is permuted identically so it cancels in P@V). pos -> key:
__device__ __forceinline__ int pos2key(int p) {
    return (p & 32) + (((p & 1) << 4) | ((p & 31) >> 1));
}

// ---------------------------------------------------------------------------
// Pre-pass 2: V fp32 [bh][L][D] -> f16 transposed+permuted Vt [bh][D][L]
// Vt[bh][d][kb + p] = V[bh][kb + pos2key(p)][d]
// ---------------------------------------------------------------------------
__global__ __launch_bounds__(256)
void transp_v(const float* __restrict__ V, _Float16* __restrict__ Vt) {
    __shared__ _Float16 tile[64][66];
    const int bh = blockIdx.x >> 5;
    const int kb = (blockIdx.x & 31) << 6;
    const int t  = threadIdx.x;

    const int kr  = t >> 2;
    const int dsg = (t & 3) << 4;
    const float* src = V + ((size_t)bh * L_ + kb + kr) * D_ + dsg;
    #pragma unroll
    for (int j = 0; j < 16; j += 4) {
        float4 f = *(const float4*)(src + j);
        tile[kr][dsg + j + 0] = (_Float16)f.x;
        tile[kr][dsg + j + 1] = (_Float16)f.y;
        tile[kr][dsg + j + 2] = (_Float16)f.z;
        tile[kr][dsg + j + 3] = (_Float16)f.w;
    }
    __syncthreads();

    const int dr = t >> 2;
    const int ps = (t & 3) << 4;
    v16h o;
    #pragma unroll
    for (int j = 0; j < 16; ++j)
        o[j] = tile[pos2key(ps + j)][dr];
    *(v16h*)(Vt + ((size_t)bh * D_ + dr) * L_ + kb + ps) = o;
}

// ---------------------------------------------------------------------------
// Main flash-attention kernel. 4 waves/block, 16 q-rows per wave, 64 keys/iter.
// 18 v_wmma per iteration: 8 (QK^T, mask folded into C) + 2 (rowsum) + 8 (P.V)
// ---------------------------------------------------------------------------
template <bool PRE>
__global__ __launch_bounds__(128)
void mha_fwd(const float* __restrict__ Q, const float* __restrict__ K32,
             const float* __restrict__ V32, const _Float16* __restrict__ K16,
             const _Float16* __restrict__ V16t, const float* __restrict__ Mask,
             float* __restrict__ Out) {
    __shared__ __align__(16) _Float16 pbuf[4][16][72];

    const int lane = threadIdx.x & 31;
    const int wave = threadIdx.x >> 5;
    const int ln   = lane & 15;
    const int hi   = lane >> 4;

    const int bh    = blockIdx.x >> 5;
    const int qt64  = blockIdx.x & 31;
    const int b     = bh >> 3;
    const int hd    = bh & 7;
    const int qbase = qt64 * 64 + wave * 16;

    const float*    Qh   = Q    + (size_t)bh * L_ * D_;
    const float*    K32h = K32  + (size_t)bh * L_ * D_;
    const float*    V32h = V32  + (size_t)bh * L_ * D_;
    const _Float16* K16h = K16  + (size_t)bh * L_ * D_;
    const _Float16* Vth  = V16t + (size_t)bh * D_ * L_;

    // ---- Q A-fragments (f32 source; converted once) ----
    v16h a0, a1;
    {
        const float* qr = Qh + (size_t)(qbase + ln) * D_;
        #pragma unroll
        for (int c = 0; c < 2; ++c) {
            const float* p0 = qr + c * 32 + hi * 16;
            const float* p1 = p0 + 8;
            v16h a;
            #pragma unroll
            for (int i = 0; i < 8; ++i) {
                a[i]     = (_Float16)p0[i];
                a[i + 8] = (_Float16)p1[i];
            }
            if (c == 0) a0 = a; else a1 = a;
        }
    }

    v16h ones;
    #pragma unroll
    for (int i = 0; i < 16; ++i) ones[i] = (_Float16)1.0f;

    float mrow[8], lrow[8];
    v8f acc[4];
    #pragma unroll
    for (int r = 0; r < 8; ++r) { mrow[r] = -3.0e38f; lrow[r] = 0.0f; }
    #pragma unroll
    for (int dt = 0; dt < 4; ++dt) acc[dt] = (v8f){0,0,0,0,0,0,0,0};

    for (int kb = 0; kb < L_; kb += 64) {
        // ---- scores: 4 key sub-tiles; mask tile fed as WMMA C operand ----
        v8f s[4];
        #pragma unroll
        for (int t = 0; t < 4; ++t) {
            v16h k0, k1;
            if constexpr (PRE) {
                const _Float16* krow = K16h + (size_t)(kb + t * 16 + ln) * D_;
                k0 = *(const v16h*)(krow + hi * 16);
                k1 = *(const v16h*)(krow + 32 + hi * 16);
            } else {
                const float* krow = K32h + (size_t)(kb + t * 16 + ln) * D_;
                #pragma unroll
                for (int i = 0; i < 16; ++i) {
                    k0[i] = (_Float16)krow[hi * 16 + i];
                    k1[i] = (_Float16)krow[32 + hi * 16 + i];
                }
            }
            v8f mt;  // mask tile in C layout: (row r+hi*8, col ln)
            #pragma unroll
            for (int r = 0; r < 8; ++r)
                mt[r] = Mask[(size_t)(qbase + r + hi * 8) * L_ + kb + t * 16 + ln];

            s[t] = __builtin_amdgcn_wmma_f32_16x16x32_f16(false, a0, false, k0,
                                                          (short)0, mt, false, false);
            s[t] = __builtin_amdgcn_wmma_f32_16x16x32_f16(false, a1, false, k1,
                                                          (short)0, s[t], false, false);
        }

        // ---- row max: in-lane over 4 tiles, then packed-f16 butterfly ----
        float xr[8];
        #pragma unroll
        for (int r = 0; r < 8; ++r)
            xr[r] = fmaxf(fmaxf(s[0][r], s[1][r]), fmaxf(s[2][r], s[3][r]));

        // pack row pairs to f16x2; reduce across the 16-lane group with
        // v_pk_max_num_f16. An approximate (f16-rounded) max is fine: softmax
        // is invariant to the subtracted constant.
        union PU { v2h h; unsigned u; };
        unsigned pk[4];
        #pragma unroll
        for (int j = 0; j < 4; ++j) {
            PU p;
            p.h[0] = (_Float16)xr[2 * j];
            p.h[1] = (_Float16)xr[2 * j + 1];
            pk[j] = p.u;
        }
        #pragma unroll
        for (int off = 1; off < 16; off <<= 1) {
            #pragma unroll
            for (int j = 0; j < 4; ++j) {
                unsigned o = (unsigned)__shfl_xor((int)pk[j], off, 32);
                asm("v_pk_max_num_f16 %0, %1, %2"
                    : "=v"(pk[j]) : "v"(pk[j]), "v"(o));
            }
        }
        #pragma unroll
        for (int j = 0; j < 4; ++j) {
            PU p; p.u = pk[j];
            xr[2 * j]     = (float)p.h[0];
            xr[2 * j + 1] = (float)p.h[1];
        }

        // ---- online softmax update + P to LDS (permuted, packed b32) ----
        float sc[8];
        #pragma unroll
        for (int r = 0; r < 8; ++r) {
            const float mnew = fmaxf(mrow[r], xr[r]);
            sc[r] = __expf(mrow[r] - mnew);
            mrow[r] = mnew;

            const float p0 = __expf(s[0][r] - mnew);
            const float p1 = __expf(s[1][r] - mnew);
            const float p2 = __expf(s[2][r] - mnew);
            const float p3 = __expf(s[3][r] - mnew);

            const int row = r + hi * 8;
            v2h e0; e0[0] = (_Float16)p0; e0[1] = (_Float16)p1;
            v2h e1; e1[0] = (_Float16)p2; e1[1] = (_Float16)p3;
            *(v2h*)(&pbuf[wave][row][ln * 2])      = e0;
            *(v2h*)(&pbuf[wave][row][32 + ln * 2]) = e1;
        }
        __syncthreads();

        // ---- P A-fragments (two K=32 chunks over the 64 permuted keys) ----
        v16h pa0, pa1;
        {
            const _Float16* pr = &pbuf[wave][ln][0];
            v8h c0 = *(const v8h*)(pr + hi * 16);
            v8h c1 = *(const v8h*)(pr + hi * 16 + 8);
            v8h c2 = *(const v8h*)(pr + 32 + hi * 16);
            v8h c3 = *(const v8h*)(pr + 32 + hi * 16 + 8);
            #pragma unroll
            for (int i = 0; i < 8; ++i) {
                pa0[i] = c0[i]; pa0[i + 8] = c1[i];
                pa1[i] = c2[i]; pa1[i + 8] = c3[i];
            }
        }

        // ---- row sums via WMMA: P x ones ----
        {
            const v8f z = (v8f){0,0,0,0,0,0,0,0};
            v8f ls = __builtin_amdgcn_wmma_f32_16x16x32_f16(false, pa0, false, ones,
                                                            (short)0, z, false, false);
            ls     = __builtin_amdgcn_wmma_f32_16x16x32_f16(false, pa1, false, ones,
                                                            (short)0, ls, false, false);
            #pragma unroll
            for (int r = 0; r < 8; ++r)
                lrow[r] = lrow[r] * sc[r] + ls[r];
        }
        #pragma unroll
        for (int dt = 0; dt < 4; ++dt)
            #pragma unroll
            for (int r = 0; r < 8; ++r)
                acc[dt][r] *= sc[r];

        // ---- O += P @ V ----
        #pragma unroll
        for (int cc = 0; cc < 2; ++cc) {
            const v16h pa = cc ? pa1 : pa0;
            #pragma unroll
            for (int dt = 0; dt < 4; ++dt) {
                v16h vb;
                if constexpr (PRE) {
                    vb = *(const v16h*)(Vth + (size_t)(dt * 16 + ln) * L_ +
                                        kb + cc * 32 + hi * 16);
                } else {
                    #pragma unroll
                    for (int i = 0; i < 16; ++i) {
                        const int p = cc * 32 + hi * 16 + i;
                        vb[i] = (_Float16)V32h[(size_t)(kb + pos2key(p)) * D_ +
                                               dt * 16 + ln];
                    }
                }
                acc[dt] = __builtin_amdgcn_wmma_f32_16x16x32_f16(false, pa, false, vb,
                                                                 (short)0, acc[dt],
                                                                 false, false);
            }
        }
    }

    // ---- epilogue: normalize, permute to [B, L, H*D], store fp32 ----
    #pragma unroll
    for (int r = 0; r < 8; ++r) {
        const int qrow = qbase + r + hi * 8;
        const float inv = 1.0f / lrow[r];
        float* op = Out + ((size_t)b * L_ + qrow) * MODEL_ + hd * D_;
        #pragma unroll
        for (int dt = 0; dt < 4; ++dt)
            op[dt * 16 + ln] = acc[dt][r] * inv;
    }
}

extern "C" void kernel_launch(void* const* d_in, const int* in_sizes, int n_in,
                              void* d_out, int out_size, void* d_ws, size_t ws_size,
                              hipStream_t stream) {
    const float* q    = (const float*)d_in[0];
    const float* k    = (const float*)d_in[1];
    const float* v    = (const float*)d_in[2];
    const float* mask = (const float*)d_in[3];
    float* out = (float*)d_out;

    const size_t KN   = (size_t)BH_ * L_ * D_;      // 4M elements
    const size_t need = KN * sizeof(_Float16) * 2;  // K16 + Vt16 = 16 MB
    const int blocks = BH_ * (L_ / 64);             // 1024

    if (ws_size >= need) {
        _Float16* k16  = (_Float16*)d_ws;
        _Float16* v16t = k16 + KN;
        conv_f16<<<(int)(KN / (8 * 256)), 256, 0, stream>>>(k, k16);
        transp_v<<<BH_ * (L_ / 64), 256, 0, stream>>>(v, v16t);
        mha_fwd<true><<<blocks, 128, 0, stream>>>(q, k, v, k16, v16t, mask, out);
    } else {
        mha_fwd<false><<<blocks, 128, 0, stream>>>(q, k, v, nullptr, nullptr,
                                                   mask, out);
    }
}
